// EquivariantAttention_54494545051800
// MI455X (gfx1250) — matrix operand order repaired and checked by reading, not compiled, not run on hardware
//
#include <hip/hip_runtime.h>
#include <hip/hip_bf16.h>

// ---------------- problem dims (compile-time constants) ----------------
constexpr int kD  = 512;            // model dim
constexpr int kB  = 2;              // batch
constexpr int kL  = 2048;           // sequence
constexpr int kH  = 8;              // heads
constexpr int kHD = 64;             // head dim
constexpr int kBL = kB * kL;        // 4096 flattened rows
constexpr int kBH = kB * kH;        // 16 (b,h) pairs

typedef __attribute__((ext_vector_type(16))) _Float16 v16h;
typedef __attribute__((ext_vector_type(8)))  float    v8f;

union FragH {
  v16h  h;
  uint4 q[2];
};

// ---- CDNA5 async global->LDS copy (ASYNCcnt-tracked, no VGPR staging) ----
typedef __attribute__((address_space(3))) void lds_void;

__device__ __forceinline__ unsigned lds_addr32(void* p) {
  // generic -> LDS address space -> 32-bit LDS byte address
  return (unsigned)(size_t)(lds_void*)p;
}

__device__ __forceinline__ void async_ld_b128(void* lds_dst, const void* gsrc) {
  asm volatile("global_load_async_to_lds_b128 %0, %1, off"
               :: "v"(lds_addr32(lds_dst)), "v"(gsrc)
               : "memory");
}

__device__ __forceinline__ void async_wait0() {
  asm volatile("s_wait_asynccnt 0x0" ::: "memory");
}

// A-operand (16x32 f16, MxK): lane(0-15)=M, lanes 16-31 same M other K-half.
// VGPR0-3: K = half*8 .. half*8+7 ; VGPR4-7: K = 16+half*8 .. 16+half*8+7
__device__ __forceinline__ FragH frag_a_ld(const _Float16* base, int ld) {
  const int lane = threadIdx.x & 31;
  const int m  = lane & 15;
  const int hf = lane >> 4;
  const _Float16* p = base + m * ld + hf * 8;
  FragH f;
  f.q[0] = *(const uint4*)(p);
  f.q[1] = *(const uint4*)(p + 16);
  return f;
}

// B-operand (32x16 f16, KxN): lane(0-15)=N; VGPR0-3: K = half*8 .. half*8+7 ; VGPR4-7: K = 16+half*8 .. 16+half*8+7
__device__ __forceinline__ FragH frag_b_ld(const _Float16* base, int ld) {
  const int lane = threadIdx.x & 31;
  const int n  = lane & 15;
  const int hf = lane >> 4;
  const _Float16* p = base + n * ld + hf * 8;
  FragH f;
  f.q[0] = *(const uint4*)(p);
  f.q[1] = *(const uint4*)(p + 16);
  return f;
}

__device__ __forceinline__ v8f wmma_f16f32(FragH a, FragH b, v8f c) {
  return __builtin_amdgcn_wmma_f32_16x16x32_f16(false, a.h, false, b.h,
                                                (short)0, c, false, false);
}

// ---------------- kernel 1: W = I + (A - A^T) from packed upper-tri coefs ---
__global__ void build_w_f16(const float* __restrict__ cq, const float* __restrict__ ck,
                            const float* __restrict__ cv, const float* __restrict__ co,
                            _Float16* __restrict__ w) {
  const int idx = blockIdx.x * blockDim.x + threadIdx.x;      // 0 .. D*D-1
  const int z   = blockIdx.y;                                 // matrix select
  const float* c = (z == 0) ? cq : (z == 1) ? ck : (z == 2) ? cv : co;
  const int i = idx >> 9;          // row
  const int j = idx & (kD - 1);    // col
  float v;
  if (i == j) {
    v = 1.0f;
  } else if (i < j) {
    v =  c[(size_t)i * (kD - 1) - (size_t)i * (i - 1) / 2 + (j - i - 1)];
  } else {
    v = -c[(size_t)j * (kD - 1) - (size_t)j * (j - 1) / 2 + (i - j - 1)];
  }
  w[(size_t)z * kD * kD + idx] = (_Float16)v;
}

// ---------------- kernel 2: f32 -> f16 bulk convert ----------------
__global__ void cvt_f32_to_f16(const float* __restrict__ x, _Float16* __restrict__ y, int n) {
  const int i = blockIdx.x * blockDim.x + threadIdx.x;
  if (i < n) y[i] = (_Float16)x[i];
}

// ---------------- kernel 3: WMMA GEMM  C = A @ W^T + bias ----------------
// A: [kBL, kD] f16 row-major.  W: [N, K] f16 row-major (so B[k][n] = W[n][k]).
// Double-buffered LDS, async global->LDS staging overlapped with WMMA.
// mode 0: f16 head-split [B,H,L,HD].  mode 1: f32 flat [kBL,kD].
// mode 2: f16 per-head transposed [B,H,HD,L] (for attention V operand).
__global__ __launch_bounds__(128)
void gemm_proj(const _Float16* __restrict__ A, const _Float16* __restrict__ W,
               const float* __restrict__ bias, _Float16* __restrict__ dst_h,
               float* __restrict__ dst_f, int mode) {
  __shared__ __align__(16) _Float16 As[2][64][32];
  __shared__ __align__(16) _Float16 Bs[2][64][32];
  const int tid  = threadIdx.x;
  const int wv   = tid >> 5;
  const int lane = tid & 31;
  const int hf   = lane >> 4;
  const int nl   = lane & 15;
  const int m0   = blockIdx.x * 64;
  const int n0   = blockIdx.y * 64;

  // this thread's two (row, col-chunk) slots of the 64x32 tiles
  const int u0 = tid * 2;
  const int r0 = u0 >> 2,       c0 = (u0 & 3) * 8;
  const int r1 = (u0 + 1) >> 2, c1 = ((u0 + 1) & 3) * 8;

  v8f acc[4] = {};

  // prologue: stage tile k0=0 into buffer 0
  async_ld_b128(&As[0][r0][c0], A + (size_t)(m0 + r0) * kD + c0);
  async_ld_b128(&Bs[0][r0][c0], W + (size_t)(n0 + r0) * kD + c0);
  async_ld_b128(&As[0][r1][c1], A + (size_t)(m0 + r1) * kD + c1);
  async_ld_b128(&Bs[0][r1][c1], W + (size_t)(n0 + r1) * kD + c1);
  async_wait0();
  __syncthreads();

  int pb = 0;
  for (int k0 = 0; k0 < kD; k0 += 32) {
    // issue async stage of next tile into the other buffer
    if (k0 + 32 < kD) {
      const int kn = k0 + 32, nb = pb ^ 1;
      async_ld_b128(&As[nb][r0][c0], A + (size_t)(m0 + r0) * kD + kn + c0);
      async_ld_b128(&Bs[nb][r0][c0], W + (size_t)(n0 + r0) * kD + kn + c0);
      async_ld_b128(&As[nb][r1][c1], A + (size_t)(m0 + r1) * kD + kn + c1);
      async_ld_b128(&Bs[nb][r1][c1], W + (size_t)(n0 + r1) * kD + kn + c1);
    }

    // compute on current buffer while the async DMA lands
    FragH a = frag_a_ld(&As[pb][wv * 16][0], 32);
#pragma unroll
    for (int t = 0; t < 4; ++t) {
      FragH b = frag_b_ld(&Bs[pb][t * 16][0], 32);
      acc[t] = wmma_f16f32(a, b, acc[t]);
    }

    async_wait0();        // my async stores into LDS have landed
    __syncthreads();      // everyone done computing & staging
    pb ^= 1;
  }

  // epilogue: C layout — VGPR r: M = r + hf*8, N = nl
#pragma unroll
  for (int t = 0; t < 4; ++t) {
#pragma unroll
    for (int r = 0; r < 8; ++r) {
      const int gm = m0 + wv * 16 + r + hf * 8;
      const int gn = n0 + t * 16 + nl;
      const float val = acc[t][r] + bias[gn];
      if (mode == 0) {
        const int b = gm >> 11, l = gm & (kL - 1);
        const int h = gn >> 6,  d = gn & (kHD - 1);
        dst_h[(((size_t)(b * kH + h)) * kL + l) * kHD + d] = (_Float16)val;
      } else if (mode == 2) {
        const int b = gm >> 11, l = gm & (kL - 1);
        const int h = gn >> 6,  d = gn & (kHD - 1);
        dst_h[(((size_t)(b * kH + h)) * kHD + d) * kL + l] = (_Float16)val;
      } else {
        dst_f[(size_t)gm * kD + gn] = val;
      }
    }
  }
}

// ---------------- kernel 4: invariants  [norm, t @ basis^T][:HD] ----------------
__global__ void invariants_k(const _Float16* __restrict__ T, const float* __restrict__ basis,
                             _Float16* __restrict__ O) {
  __shared__ float q[kHD];
  const size_t row = blockIdx.x;
  const int n = threadIdx.x;
  q[n] = (float)T[row * kHD + n];
  __syncthreads();
  float r;
  if (n == 0) {
    float s = 0.f;
#pragma unroll
    for (int d = 0; d < kHD; ++d) s += q[d] * q[d];
    r = sqrtf(s);
  } else {
    const float* bs = basis + (size_t)(n - 1) * kHD;
    float s = 0.f;
#pragma unroll
    for (int d = 0; d < kHD; ++d) s += q[d] * bs[d];
    r = s;
  }
  O[row * kHD + n] = (_Float16)r;
}

// ---------------- kernel 5: causal flash attention (WMMA) ----------------
// Qi/Ki: [BH,L,HD] f16.  Vt: [BH,HD,L] f16 (pre-transposed by gemm_proj mode 2).
// grid: (L/64 query tiles, B*H). block: 128 threads (4 waves, 16 rows each).
__global__ __launch_bounds__(128)
void flash_attn(const _Float16* __restrict__ Qi, const _Float16* __restrict__ Ki,
                const _Float16* __restrict__ Vt, _Float16* __restrict__ Oh) {
  __shared__ __align__(16) _Float16 qs[64][64];   // Qi tile
  __shared__ __align__(16) _Float16 ks[64][64];   // Ki tile (row-major: B for QK^T)
  __shared__ __align__(16) _Float16 vt[64][64];   // V tile transposed: vt[d][m]
  __shared__ __align__(16) float    sf[64][64];   // scores f32
  __shared__ __align__(16) _Float16 ph[64][64];   // exp(P) f16 (A for PV)
  __shared__ float m_s[64], l_s[64], resc[64];

  const int tid  = threadIdx.x;
  const int wv   = tid >> 5;
  const int lane = tid & 31;
  const int hf   = lane >> 4;
  const int nl   = lane & 15;
  const int q0   = blockIdx.x * 64;
  const int bh   = blockIdx.y;
  const size_t base  = (size_t)bh * kL * kHD;   // Qi/Ki head base
  const size_t baseV = (size_t)bh * kHD * kL;   // Vt head base

  // Qi tile: contiguous, 512 b128 chunks, async into LDS
  for (int u = tid; u < 512; u += 128)
    async_ld_b128((_Float16*)qs + u * 8, Qi + base + (size_t)q0 * kHD + u * 8);
  if (tid < 64) { m_s[tid] = -1e30f; l_s[tid] = 0.f; }
  async_wait0();
  __syncthreads();

  FragH aq0 = frag_a_ld(&qs[wv * 16][0], 64);
  FragH aq1 = frag_a_ld(&qs[wv * 16][32], 64);
  v8f o[4] = {};

  const int jmax = q0 / 64;   // causal: keys 0..q0+63
  for (int j = 0; j <= jmax; ++j) {
    const int k0 = j * 64;
    // Ki tile: contiguous rows -> flat async copy
    for (int u = tid; u < 512; u += 128)
      async_ld_b128((_Float16*)ks + u * 8, Ki + base + (size_t)k0 * kHD + u * 8);
    // V tile: Vt rows are length-L; take 64-wide slice per head-dim row
    for (int u = tid; u < 512; u += 128) {
      const int d = u >> 3, c8 = (u & 7) * 8;
      async_ld_b128(&vt[d][c8], Vt + baseV + (size_t)d * kL + k0 + c8);
    }
    async_wait0();
    __syncthreads();

    // S = (Qi @ Ki^T) / 8 with causal mask -> sf
#pragma unroll
    for (int t = 0; t < 4; ++t) {
      v8f c = {};
      FragH b0 = frag_b_ld(&ks[t * 16][0], 64);
      FragH b1 = frag_b_ld(&ks[t * 16][32], 64);
      c = wmma_f16f32(aq0, b0, c);
      c = wmma_f16f32(aq1, b1, c);
      const int kn = k0 + t * 16 + nl;
#pragma unroll
      for (int r = 0; r < 8; ++r) {
        const int qm = q0 + wv * 16 + r + hf * 8;
        float s = c[r] * 0.125f;              // 1/sqrt(HD)
        if (kn > qm) s = -1e30f;              // causal mask
        sf[wv * 16 + r + hf * 8][t * 16 + nl] = s;
      }
    }
    __syncthreads();

    // online softmax: 2 threads per row (adjacent lanes), 32 cols each
    {
      const int row = tid >> 1, hc = (tid & 1) * 32;
      float mx = -1e30f;
#pragma unroll
      for (int c = 0; c < 32; ++c) mx = fmaxf(mx, sf[row][hc + c]);
      mx = fmaxf(mx, __shfl_xor(mx, 1));
      const float mold = m_s[row];
      const float mnew = fmaxf(mold, mx);
      float sum = 0.f;
#pragma unroll
      for (int c = 0; c < 32; ++c) {
        const float p = __expf(sf[row][hc + c] - mnew);
        ph[row][hc + c] = (_Float16)p;
        sum += p;
      }
      sum += __shfl_xor(sum, 1);
      if ((tid & 1) == 0) {
        const float f = __expf(mold - mnew);
        resc[row] = f;
        m_s[row]  = mnew;
        l_s[row]  = l_s[row] * f + sum;
      }
    }
    __syncthreads();

    // O = O * resc + P @ V
    {
      float fr[8];
#pragma unroll
      for (int r = 0; r < 8; ++r) fr[r] = resc[wv * 16 + r + hf * 8];
      FragH ap0 = frag_a_ld(&ph[wv * 16][0], 64);
      FragH ap1 = frag_a_ld(&ph[wv * 16][32], 64);
#pragma unroll
      for (int t = 0; t < 4; ++t) {
#pragma unroll
        for (int r = 0; r < 8; ++r) o[t][r] *= fr[r];
        FragH bv0 = frag_b_ld(&vt[t * 16][0], 64);
        FragH bv1 = frag_b_ld(&vt[t * 16][32], 64);
        o[t] = wmma_f16f32(ap0, bv0, o[t]);
        o[t] = wmma_f16f32(ap1, bv1, o[t]);
      }
    }
    __syncthreads();
  }

  // normalize by l and store back as [B, L, D] f16 (merge heads)
  const int b = bh / kH, h = bh % kH;
#pragma unroll
  for (int t = 0; t < 4; ++t) {
#pragma unroll
    for (int r = 0; r < 8; ++r) {
      const int qm  = q0 + wv * 16 + r + hf * 8;
      const float v = o[t][r] / l_s[qm - q0];
      Oh[((size_t)b * kL + qm) * kD + h * kHD + t * 16 + nl] = (_Float16)v;
    }
  }
}

// ---------------- launcher ----------------
extern "C" void kernel_launch(void* const* d_in, const int* in_sizes, int n_in,
                              void* d_out, int out_size, void* d_ws, size_t ws_size,
                              hipStream_t stream) {
  (void)in_sizes; (void)n_in; (void)out_size; (void)ws_size;
  const float* x       = (const float*)d_in[0];
  /* d_in[1] = mask: causal, handled analytically */
  const float* coef_q  = (const float*)d_in[2];
  const float* coef_k  = (const float*)d_in[3];
  const float* coef_v  = (const float*)d_in[4];
  const float* coef_o  = (const float*)d_in[5];
  const float* bias_q  = (const float*)d_in[6];
  const float* bias_k  = (const float*)d_in[7];
  const float* bias_v  = (const float*)d_in[8];
  const float* bias_o  = (const float*)d_in[9];
  const float* basis_q = (const float*)d_in[10];
  const float* basis_k = (const float*)d_in[11];
  float* out = (float*)d_out;

  // workspace layout (bytes): 4 W (2 MiB) + xh (4) + Qh/Kh/Vt (12) + Qi/Ki (8) + Oh (4) = 30 MiB
  char* ws = (char*)d_ws;
  size_t off = 0;
  _Float16* W4 = (_Float16*)(ws + off); off += (size_t)4 * kD * kD * sizeof(_Float16);
  _Float16* xh = (_Float16*)(ws + off); off += (size_t)kBL * kD * sizeof(_Float16);
  _Float16* Qh = (_Float16*)(ws + off); off += (size_t)kBH * kL * kHD * sizeof(_Float16);
  _Float16* Kh = (_Float16*)(ws + off); off += (size_t)kBH * kL * kHD * sizeof(_Float16);
  _Float16* Vt = (_Float16*)(ws + off); off += (size_t)kBH * kHD * kL * sizeof(_Float16);
  _Float16* Qi = (_Float16*)(ws + off); off += (size_t)kBH * kL * kHD * sizeof(_Float16);
  _Float16* Ki = (_Float16*)(ws + off); off += (size_t)kBH * kL * kHD * sizeof(_Float16);
  _Float16* Oh = (_Float16*)(ws + off); off += (size_t)kBL * kD * sizeof(_Float16);

  const _Float16* Wq = W4 + (size_t)0 * kD * kD;
  const _Float16* Wk = W4 + (size_t)1 * kD * kD;
  const _Float16* Wv = W4 + (size_t)2 * kD * kD;
  const _Float16* Wo = W4 + (size_t)3 * kD * kD;

  // 1) build weights, convert x
  build_w_f16<<<dim3(kD * kD / 256, 4), 256, 0, stream>>>(coef_q, coef_k, coef_v, coef_o, W4);
  cvt_f32_to_f16<<<(kBL * kD) / 256, 256, 0, stream>>>(x, xh, kBL * kD);

  // 2) Q/K/V projections (Q,K head-split f16; V head-split transposed f16)
  dim3 ggrid(kBL / 64, kD / 64);
  gemm_proj<<<ggrid, 128, 0, stream>>>(xh, Wq, bias_q, Qh, nullptr, 0);
  gemm_proj<<<ggrid, 128, 0, stream>>>(xh, Wk, bias_k, Kh, nullptr, 0);
  gemm_proj<<<ggrid, 128, 0, stream>>>(xh, Wv, bias_v, Vt, nullptr, 2);

  // 3) invariants
  invariants_k<<<kBH * kL, kHD, 0, stream>>>(Qh, basis_q, Qi);
  invariants_k<<<kBH * kL, kHD, 0, stream>>>(Kh, basis_k, Ki);

  // 4) causal flash attention
  flash_attn<<<dim3(kL / 64, kBH), 128, 0, stream>>>(Qi, Ki, Vt, Oh);

  // 5) output projection (f32 flat out)
  gemm_proj<<<ggrid, 128, 0, stream>>>(Oh, Wo, bias_o, nullptr, out, 1);
}
